// GGNNLayer_80221399155535
// MI455X (gfx1250) — matrix operand order repaired and checked by reading, not compiled, not run on hardware
//
#include <hip/hip_runtime.h>
#include <hip/hip_bf16.h>

typedef __attribute__((ext_vector_type(16))) __bf16 v16bf;
typedef __attribute__((ext_vector_type(8)))  __bf16 v8bf;
typedef __attribute__((ext_vector_type(8)))  float  v8f;

#define DD 128
#define LDS_STRIDE 136   // bf16 elems; 272B rows = 17 uint4 -> 16B aligned, bank-staggered

// ---------------- WMMA helpers ----------------

__device__ __forceinline__ v8f wmma_bf16(v16bf a, v16bf b, v8f c) {
  return __builtin_amdgcn_wmma_f32_16x16x32_bf16(
      /*neg_a=*/false, a, /*neg_b=*/false, b,
      /*c_mod=*/(short)0, c, /*reuse_a=*/false, /*reuse_b=*/false);
}

// A fragment (16x32 bf16). rowptr = &A[row][k0] (row for lane&15), g = lane>>4.
// element e -> K=k0+g*16+e
__device__ __forceinline__ v16bf make_afrag(const __bf16* rowptr, int g) {
  v8bf lo = *(const v8bf*)(rowptr + g * 16);
  v8bf hi = *(const v8bf*)(rowptr + g * 16 + 8);
  v16bf r;
#pragma unroll
  for (int i = 0; i < 8; ++i) { r[i] = lo[i]; r[i + 8] = hi[i]; }
  return r;
}

// B fragment (32x16 bf16) from transposed weights Bt[outcol][128] (global).
// n = this lane's output column. element e -> K = k0 + g*16 + e.
__device__ __forceinline__ v16bf make_bfrag(const __bf16* __restrict__ Bt, int n,
                                            int k0, int g) {
  const __bf16* p = Bt + (size_t)n * DD + k0 + g * 16;
  v8bf lo = *(const v8bf*)p;
  v8bf hi = *(const v8bf*)(p + 8);
  v16bf r;
#pragma unroll
  for (int i = 0; i < 8; ++i) { r[i] = lo[i]; r[i + 8] = hi[i]; }
  return r;
}

// B fragment from LDS-staged weights with padded stride
__device__ __forceinline__ v16bf make_bfrag_lds(const __bf16* sW, int n, int k0, int g) {
  const __bf16* p = sW + n * LDS_STRIDE + k0 + g * 16;
  v8bf lo = *(const v8bf*)p;
  v8bf hi = *(const v8bf*)(p + 8);
  v16bf r;
#pragma unroll
  for (int i = 0; i < 8; ++i) { r[i] = lo[i]; r[i + 8] = hi[i]; }
  return r;
}

__device__ __forceinline__ float sigmoidf(float x) { return 1.0f / (1.0f + __expf(-x)); }

// ---------------- prep kernels ----------------

__global__ void k_cvt_bf16(const float* __restrict__ in, __bf16* __restrict__ out, int n) {
  int i = blockIdx.x * blockDim.x + threadIdx.x;
  if (i < n) out[i] = (__bf16)in[i];
}

// W is [128, C] row-major; Bt is [C, 128] (transposed, bf16)
__global__ void k_transpose_cvt(const float* __restrict__ W, __bf16* __restrict__ Bt, int C) {
  int i = blockIdx.x * blockDim.x + threadIdx.x;
  if (i < DD * C) {
    int k = i / C, c = i - k * C;
    Bt[(size_t)c * DD + k] = (__bf16)W[i];
  }
}

__global__ void k_zero(float* __restrict__ p, int n) {
  int i = blockIdx.x * blockDim.x + threadIdx.x;
  if (i < n) p[i] = 0.0f;
}

// ---------------- msg_net: X_msg = (X@W1+b1)@W2+b2 ----------------

__global__ void __launch_bounds__(256) k_msg(const __bf16* __restrict__ Xh,
                                             const __bf16* __restrict__ W1t,
                                             const __bf16* __restrict__ W2t,
                                             const float* __restrict__ b1,
                                             const float* __restrict__ b2,
                                             float* __restrict__ Xmsg, int n_rows) {
  __shared__ __align__(16) __bf16 sW1[128 * LDS_STRIDE];
  __shared__ __align__(16) __bf16 sW2[128 * LDS_STRIDE];
  __shared__ __align__(16) __bf16 T[128 * LDS_STRIDE];

  const int tid = threadIdx.x;
  const int w = tid >> 5, lane = tid & 31;
  const int g = lane >> 4, nl = lane & 15;
  const int row0 = blockIdx.x * 128;

  // cooperative stage of both weight matrices into padded LDS (uint4 = 8 bf16)
  {
    uint4* d1 = (uint4*)sW1;
    uint4* d2 = (uint4*)sW2;
    const uint4* s1 = (const uint4*)W1t;
    const uint4* s2 = (const uint4*)W2t;
#pragma unroll 2
    for (int i = tid; i < 128 * 16; i += 256) {
      int r = i >> 4, c4 = i & 15;
      d1[r * 17 + c4] = s1[r * 16 + c4];
      d2[r * 17 + c4] = s2[r * 16 + c4];
    }
  }
  __syncthreads();

  // ---- GEMM1: rows [row0+w*16, +16), all 128 cols; A from global bf16 X ----
  v8f acc[8];
  const v8f vzero = {0.f, 0.f, 0.f, 0.f, 0.f, 0.f, 0.f, 0.f};
#pragma unroll
  for (int ct = 0; ct < 8; ++ct) acc[ct] = vzero;

  int arow = row0 + w * 16 + nl;
  if (arow >= n_rows) arow = n_rows - 1;           // clamp; extra rows discarded later
  const __bf16* arow_p = Xh + (size_t)arow * DD;

#pragma unroll
  for (int kk = 0; kk < 4; ++kk) {
    v16bf a = make_afrag(arow_p + kk * 32, g);
    v16bf bf[4];
#pragma unroll
    for (int h = 0; h < 2; ++h) {                  // groups of 4: batch loads, then WMMAs
#pragma unroll
      for (int j = 0; j < 4; ++j) bf[j] = make_bfrag_lds(sW1, (h * 4 + j) * 16 + nl, kk * 32, g);
#pragma unroll
      for (int j = 0; j < 4; ++j) acc[h * 4 + j] = wmma_bf16(a, bf[j], acc[h * 4 + j]);
    }
  }

  // +b1, convert to bf16, stage in LDS
#pragma unroll
  for (int ct = 0; ct < 8; ++ct) {
    float bias = b1[ct * 16 + nl];
#pragma unroll
    for (int i = 0; i < 8; ++i) {
      int rl = w * 16 + i + 8 * g;
      T[rl * LDS_STRIDE + ct * 16 + nl] = (__bf16)(acc[ct][i] + bias);
    }
  }
  __syncthreads();

  // ---- GEMM2: A from LDS, B = sW2, +b2, write fp32 X_msg ----
#pragma unroll
  for (int ct = 0; ct < 8; ++ct) acc[ct] = vzero;
  const __bf16* lrow_p = T + (w * 16 + nl) * LDS_STRIDE;

#pragma unroll
  for (int kk = 0; kk < 4; ++kk) {
    v16bf a = make_afrag(lrow_p + kk * 32, g);
    v16bf bf[4];
#pragma unroll
    for (int h = 0; h < 2; ++h) {
#pragma unroll
      for (int j = 0; j < 4; ++j) bf[j] = make_bfrag_lds(sW2, (h * 4 + j) * 16 + nl, kk * 32, g);
#pragma unroll
      for (int j = 0; j < 4; ++j) acc[h * 4 + j] = wmma_bf16(a, bf[j], acc[h * 4 + j]);
    }
  }

#pragma unroll
  for (int ct = 0; ct < 8; ++ct) {
    float bias = b2[ct * 16 + nl];
#pragma unroll
    for (int i = 0; i < 8; ++i) {
      int row = row0 + w * 16 + i + 8 * g;
      if (row < n_rows) Xmsg[(size_t)row * DD + ct * 16 + nl] = acc[ct][i] + bias;
    }
  }
}

// ---------------- undirected scatter-add: one wave per edge-direction ----------------

__global__ void __launch_bounds__(256) k_scatter(const float* __restrict__ Xmsg,
                                                 const int* __restrict__ ra,
                                                 const int* __restrict__ rb,
                                                 float* __restrict__ agg, int E) {
  int wid = (blockIdx.x * blockDim.x + threadIdx.x) >> 5;
  int lane = threadIdx.x & 31;
  if (wid >= 2 * E) return;
  int e = (wid < E) ? wid : wid - E;
  int src = (wid < E) ? ra[e] : rb[e];
  int dst = (wid < E) ? rb[e] : ra[e];
  const float4 v = *(const float4*)(Xmsg + (size_t)src * DD + lane * 4);
  float* p = agg + (size_t)dst * DD + lane * 4;
  atomicAdd(p + 0, v.x);
  atomicAdd(p + 1, v.y);
  atomicAdd(p + 2, v.z);
  atomicAdd(p + 3, v.w);
}

// ---------------- fused GRU: mx=agg@K+b0, mh=X@Kr+b1, gates, blend ----------------

__global__ void __launch_bounds__(256) k_gru(const float* __restrict__ X,
                                             const __bf16* __restrict__ Xh,
                                             const float* __restrict__ agg,
                                             const __bf16* __restrict__ Kat,
                                             const __bf16* __restrict__ Krt,
                                             const float* __restrict__ gbias,
                                             float* __restrict__ out, int n_rows) {
  __shared__ __align__(16) __bf16 Ag[128 * LDS_STRIDE];
  const int tid = threadIdx.x;
  const int row0 = blockIdx.x * 128;

  // stage agg rows as bf16 in LDS
  for (int t = tid; t < 128 * 128; t += 256) {
    int r = t >> 7, c = t & 127;
    int row = row0 + r;
    float v = (row < n_rows) ? agg[(size_t)row * DD + c] : 0.0f;
    Ag[r * LDS_STRIDE + c] = (__bf16)v;
  }
  __syncthreads();

  const int w = tid >> 5, lane = tid & 31;
  const int g = lane >> 4, nl = lane & 15;

  int arow = row0 + w * 16 + nl;
  if (arow >= n_rows) arow = n_rows - 1;
  const __bf16* xrow_p = Xh + (size_t)arow * DD;
  const __bf16* grow_p = Ag + (w * 16 + nl) * LDS_STRIDE;

  const v8f vzero = {0.f, 0.f, 0.f, 0.f, 0.f, 0.f, 0.f, 0.f};

  for (int c = 0; c < 8; ++c) {           // output col-tile (16 cols each)
    v8f acc6[6];                          // az, ar, ah (agg@K) ; hz, hr, hh (X@Kr)
#pragma unroll
    for (int j = 0; j < 6; ++j) acc6[j] = vzero;
    const int n = c * 16 + nl;

#pragma unroll
    for (int kk = 0; kk < 4; ++kk) {
      int k0 = kk * 32;
      v16bf aA = make_afrag(grow_p + k0, g);
      v16bf aX = make_afrag(xrow_p + k0, g);
      // batch all 6 B-fragment loads (12 x b128) before the 6 WMMAs
      v16bf bf[6];
      bf[0] = make_bfrag(Kat, n, k0, g);
      bf[1] = make_bfrag(Kat, 128 + n, k0, g);
      bf[2] = make_bfrag(Kat, 256 + n, k0, g);
      bf[3] = make_bfrag(Krt, n, k0, g);
      bf[4] = make_bfrag(Krt, 128 + n, k0, g);
      bf[5] = make_bfrag(Krt, 256 + n, k0, g);
      acc6[0] = wmma_bf16(aA, bf[0], acc6[0]);
      acc6[1] = wmma_bf16(aA, bf[1], acc6[1]);
      acc6[2] = wmma_bf16(aA, bf[2], acc6[2]);
      acc6[3] = wmma_bf16(aX, bf[3], acc6[3]);
      acc6[4] = wmma_bf16(aX, bf[4], acc6[4]);
      acc6[5] = wmma_bf16(aX, bf[5], acc6[5]);
    }

    float b_z = gbias[n],        b_r = gbias[128 + n],        b_h = gbias[256 + n];
    float rb_z = gbias[384 + n], rb_r = gbias[384 + 128 + n], rb_h = gbias[384 + 256 + n];

#pragma unroll
    for (int i = 0; i < 8; ++i) {
      int row = row0 + w * 16 + i + 8 * g;
      if (row < n_rows) {
        float z = sigmoidf(acc6[0][i] + b_z + acc6[3][i] + rb_z);
        float r = sigmoidf(acc6[1][i] + b_r + acc6[4][i] + rb_r);
        float hc = tanhf(acc6[2][i] + b_h + r * (acc6[5][i] + rb_h));
        float x = X[(size_t)row * DD + n];
        out[(size_t)row * DD + n] = z * x + (1.0f - z) * hc;
      }
    }
  }
}

// ---------------- launcher ----------------

extern "C" void kernel_launch(void* const* d_in, const int* in_sizes, int n_in,
                              void* d_out, int out_size, void* d_ws, size_t ws_size,
                              hipStream_t stream) {
  (void)n_in; (void)out_size; (void)ws_size;
  const float* X  = (const float*)d_in[0];
  const int*   ra = (const int*)d_in[1];
  const int*   rb = (const int*)d_in[2];
  const float* W1 = (const float*)d_in[3];
  const float* b1 = (const float*)d_in[4];
  const float* W2 = (const float*)d_in[5];
  const float* b2 = (const float*)d_in[6];
  const float* GK = (const float*)d_in[7];
  const float* GR = (const float*)d_in[8];
  const float* GB = (const float*)d_in[9];
  float* out = (float*)d_out;

  const int N = in_sizes[0] / DD;
  const int E = in_sizes[1];

  char* ws = (char*)d_ws;
  size_t off = 0;
  auto alloc = [&](size_t bytes) -> void* {
    void* p = (void*)(ws + off);
    off += (bytes + 255) & ~(size_t)255;
    return p;
  };
  __bf16* Xh   = (__bf16*)alloc((size_t)N * DD * sizeof(__bf16));
  __bf16* W1t  = (__bf16*)alloc((size_t)DD * DD * sizeof(__bf16));
  __bf16* W2t  = (__bf16*)alloc((size_t)DD * DD * sizeof(__bf16));
  __bf16* Kat  = (__bf16*)alloc((size_t)384 * DD * sizeof(__bf16));
  __bf16* Krt  = (__bf16*)alloc((size_t)384 * DD * sizeof(__bf16));
  float*  Xmsg = (float*)alloc((size_t)N * DD * sizeof(float));
  float*  agg  = (float*)alloc((size_t)N * DD * sizeof(float));

  const int nd = N * DD;

  // prep: bf16 X, transposed bf16 weights, zeroed agg
  k_cvt_bf16<<<(nd + 255) / 256, 256, 0, stream>>>(X, Xh, nd);
  k_transpose_cvt<<<(DD * DD + 255) / 256, 256, 0, stream>>>(W1, W1t, DD);
  k_transpose_cvt<<<(DD * DD + 255) / 256, 256, 0, stream>>>(W2, W2t, DD);
  k_transpose_cvt<<<(DD * 384 + 255) / 256, 256, 0, stream>>>(GK, Kat, 384);
  k_transpose_cvt<<<(DD * 384 + 255) / 256, 256, 0, stream>>>(GR, Krt, 384);
  k_zero<<<(nd + 255) / 256, 256, 0, stream>>>(agg, nd);

  // msg_net (two chained GEMMs, WMMA bf16, LDS-staged weights)
  const int nblk = (N + 127) / 128;
  k_msg<<<nblk, 256, 0, stream>>>(Xh, W1t, W2t, b1, b2, Xmsg, N);

  // undirected scatter-add (2E wave-tasks, 8 waves per block)
  k_scatter<<<(2 * E + 7) / 8, 256, 0, stream>>>(Xmsg, ra, rb, agg, E);

  // fused GRU (six WMMA streams per col-tile + gates)
  k_gru<<<nblk, 256, 0, stream>>>(X, Xh, agg, Kat, Krt, GB, out, N);
}
